// RecurrentGemmaRglru_11089605558882
// MI455X (gfx1250) — matrix operand order repaired and checked by reading, not compiled, not run on hardware
//
#include <hip/hip_runtime.h>
#include <hip/hip_bf16.h>

#define HNUM   10
#define BWID   256
#define SEQ    4096
#define BATCH  2
#define DMODEL 2560
#define XSTRIDE 260   // f32 row stride of an x tile: 256 data + 4 pad dwords (TDM pad)
#define XBYTES  (16 * XSTRIDE * 4)   // 16640 B per buffer
#define NCHUNK (SEQ / 16)

typedef __bf16 bf16x16 __attribute__((ext_vector_type(16)));
typedef float  f32x8   __attribute__((ext_vector_type(8)));
typedef unsigned int u32x4 __attribute__((ext_vector_type(4)));
typedef int          i32x4 __attribute__((ext_vector_type(4)));
typedef int          i32x8 __attribute__((ext_vector_type(8)));

__device__ __forceinline__ unsigned int pack_bf16(float a, float b) {
  unsigned short lo = __builtin_bit_cast(unsigned short, (__bf16)a);
  unsigned short hi = __builtin_bit_cast(unsigned short, (__bf16)b);
  return ((unsigned int)hi << 16) | lo;
}

// Rearrange f32 gate weights [H,BW,BW] (two gates) into bf16 WMMA B-fragment
// layout: [g][h][nt:16][kb:8][lane:32][8 dwords]  (2.62 MB, lives in d_ws).
__global__ void rglru_prep_weights(const float* __restrict__ wI,
                                   const float* __restrict__ wR,
                                   unsigned int* __restrict__ out) {
  int tid  = blockIdx.x * 256 + threadIdx.x;   // 0 .. 81919
  int lane = tid & 31;
  int kb   = (tid >> 5) & 7;
  int nt   = (tid >> 8) & 15;
  int gh   = tid >> 12;                        // g*HNUM + h
  int g = gh / HNUM, h = gh % HNUM;
  const float* W = (g == 0) ? wI : wR;
  int col   = nt * 16 + (lane & 15);
  int kbase = kb * 32 + ((lane >> 4) * 16);
  const float* src = W + (size_t)h * BWID * BWID + (size_t)kbase * BWID + col;
  unsigned int* dst = out + (size_t)tid * 8;
#pragma unroll
  for (int j = 0; j < 8; ++j)
    dst[j] = pack_bf16(src[(2 * j) * BWID], src[(2 * j + 1) * BWID]);
}

// TDM: DMA one 16x256 f32 tile (global row stride 2560 elems) into LDS,
// padding +4 dwords after every 256-dword row -> LDS row stride = 260 floats.
__device__ __forceinline__ void tdm_load_tile(unsigned int lds_off, const float* gptr) {
  unsigned long long ga = (unsigned long long)(size_t)gptr;
  u32x4 g0;
  g0.x = 1u;                                          // count=1, user descriptor
  g0.y = lds_off;                                     // lds_addr (bytes)
  g0.z = (unsigned int)(ga & 0xffffffffu);            // global_addr[31:0]
  g0.w = (unsigned int)((ga >> 32) & 0x01ffffffu)     // global_addr[56:32]
       | (2u << 30);                                  // type = 2 ("image")
  i32x8 g1;
  g1[0] = (int)((2u << 16)      // data_size = 4B
              | (1u << 20)      // pad_enable
              | (7u << 22)      // pad_interval: 256 dwords
              | (3u << 25));    // pad_amount: 4 dwords
  g1[1] = (int)(256u << 16);    // tensor_dim0 = 256      (bits 63:48)
  g1[2] = (int)(16u  << 16);    // tensor_dim1 = 16       (bits 95:80)
  g1[3] = (int)(256u << 16);    // tile_dim0   = 256      (bits 127:112)
  g1[4] = 16;                   // tile_dim1   = 16, tile_dim2 = 0
  g1[5] = 2560;                 // tensor_dim0_stride = 2560 elems (bits 191:160)
  g1[6] = 0;
  g1[7] = 0;
  i32x4 z4 = {0, 0, 0, 0};
#if __clang_major__ >= 23
  i32x8 z8 = {0, 0, 0, 0, 0, 0, 0, 0};
  __builtin_amdgcn_tensor_load_to_lds(g0, g1, z4, z4, z8, 0);
#else
  __builtin_amdgcn_tensor_load_to_lds(g0, g1, z4, z4, 0);
#endif
}

// Fused RG-LRU: TDM-prefetched x tiles (triple buffer, depth-2 pipeline),
// bf16 WMMA gate GEMMs (f32 acc), gating elementwise, sequential recurrence.
// Block = (column-half of a head, head, batch); 128 threads = 4 waves.
__global__ __launch_bounds__(128, 1) void rglru_fused(
    const float* __restrict__ act,
    const int*   __restrict__ pos,
    const float* __restrict__ lam,
    const float* __restrict__ biasIg,
    const float* __restrict__ biasRg,
    const unsigned int* __restrict__ wPrep,
    float* __restrict__ out)
{
  // 128 KB weights + 3 x 16.6 KB x-tile buffers + pos  (= 181 KB)
  __shared__ __align__(128) unsigned char smem[131072 + 3 * XBYTES + 64];
  unsigned int* wlds = (unsigned int*)smem;
  int* posl = (int*)(smem + 131072 + 3 * XBYTES);

  const int nhalf = blockIdx.x;             // 0..1: which 128 output cols of the head
  const int h     = blockIdx.y;
  const int b     = blockIdx.z;
  const int tid   = threadIdx.x;
  const int lane  = tid & 31;
  const int wave  = tid >> 5;

  // ---- cache both gates' weights for these 128 columns in LDS (once) ----
#pragma unroll
  for (int g = 0; g < 2; ++g) {
    const uint4* src = (const uint4*)(wPrep +
        (size_t)((g * HNUM + h) * 16 + nhalf * 8) * 2048);
    uint4* dst = (uint4*)(wlds + g * 16384);
    for (int i = tid; i < 4096; i += 128) dst[i] = src[i];
  }

  // ---- per-lane column constants (softplus(Lambda), biases) ----
  float sp[2], bI[2], bR[2];
  int cHead[2];
#pragma unroll
  for (int tI = 0; tI < 2; ++tI) {
    int ntl = wave * 2 + tI;
    int c = nhalf * 128 + ntl * 16 + (lane & 15);
    cHead[tI] = c;
    int gidx = h * BWID + c;
    sp[tI] = log1pf(__expf(lam[gidx]));
    bI[tI] = biasIg[gidx];
    bR[tI] = biasRg[gidx];
  }

  const float* actBase = act + (size_t)b * SEQ * DMODEL + (size_t)h * BWID;
  const int*   posBase = pos + (size_t)b * SEQ;
  float*       outBase = out + (size_t)b * SEQ * DMODEL + (size_t)h * BWID + nhalf * 128;

  // LDS byte offsets of the three x-tile buffers (flat-address low 32 bits)
  unsigned int xoff[3];
  float* xbuf[3];
#pragma unroll
  for (int j = 0; j < 3; ++j) {
    xbuf[j] = (float*)(smem + 131072 + j * XBYTES);
    xoff[j] = (unsigned int)(size_t)(void*)xbuf[j];
  }

  // ---- prologue: prefetch chunks 0 and 1 via TDM (wave 0 only) ----
  if (wave == 0) {
    tdm_load_tile(xoff[0], actBase);
    tdm_load_tile(xoff[1], actBase + (size_t)16 * DMODEL);
  }
  __syncthreads();   // weights staged; TDM in flight

  const int r0    = lane & 15;            // A-matrix row fed by this lane
  const int khalf = (lane >> 4) * 16;     // A-fragment K sub-offset
  const int rbase = (lane >> 4) * 8;      // C/D tile row base for this lane

  float state = 0.0f;                     // scan carry for column `tid`

  for (int i = 0; i < NCHUNK; ++i) {
    const int s0  = i * 16;
    const int cur = i % 3;
    float* xlds = xbuf[cur];
    float* aS   = xlds;                   // scratch aliases current buffer
    float* xnS  = xlds + 2048;

    // Prefetch chunk i+2, then ensure chunk i has landed (in-order TENSORcnt)
    if (wave == 0) {
      if (i + 2 < NCHUNK) {
        tdm_load_tile(xoff[(i + 2) % 3], actBase + (size_t)(s0 + 32) * DMODEL);
        __builtin_amdgcn_s_wait_tensorcnt(2);
      } else if (i + 1 < NCHUNK) {
        __builtin_amdgcn_s_wait_tensorcnt(1);
      } else {
        __builtin_amdgcn_s_wait_tensorcnt(0);
      }
    }
    if (tid < 16) posl[tid] = posBase[s0 + tid];
    __syncthreads();   // chunk i visible to all waves

    // Phase B1: build bf16 A-fragments; grab f32 x values + reset flags
    bf16x16 afrag[8];
#pragma unroll
    for (int kb = 0; kb < 8; ++kb) {
      const float* xr = xlds + r0 * XSTRIDE + kb * 32 + khalf;
#pragma unroll
      for (int j = 0; j < 8; ++j) {
        afrag[kb][j]     = (__bf16)xr[j];        // K = khalf + 0..7
        afrag[kb][8 + j] = (__bf16)xr[8 + j];    // K = khalf + 8..15
      }
    }
    float xv[2][8];
#pragma unroll
    for (int tI = 0; tI < 2; ++tI)
#pragma unroll
      for (int v = 0; v < 8; ++v)
        xv[tI][v] = xlds[(rbase + v) * XSTRIDE + cHead[tI]];
    int rstm = 0;
#pragma unroll
    for (int v = 0; v < 8; ++v) rstm |= (posl[rbase + v] == 0) << v;
    __syncthreads();    // x-tile reads done -> aS/xnS may overwrite buffer

    // Phase B2: gate GEMMs — 32 v_wmma_f32_16x16x32_bf16 per wave per chunk
    f32x8 accI[2] = { {0,0,0,0,0,0,0,0}, {0,0,0,0,0,0,0,0} };
    f32x8 accR[2] = { {0,0,0,0,0,0,0,0}, {0,0,0,0,0,0,0,0} };
#pragma unroll
    for (int kb = 0; kb < 8; ++kb) {
#pragma unroll
      for (int tI = 0; tI < 2; ++tI) {
        const int ntl = wave * 2 + tI;
        const bf16x16 bfI = *(const bf16x16*)(smem +
            (((0 * 8 + ntl) * 8 + kb) << 10) + (lane << 5));
        accI[tI] = __builtin_amdgcn_wmma_f32_16x16x32_bf16(
            false, afrag[kb], false, bfI, (short)0, accI[tI], false, false);
        const bf16x16 bfR = *(const bf16x16*)(smem +
            (((1 * 8 + ntl) * 8 + kb) << 10) + (lane << 5));
        accR[tI] = __builtin_amdgcn_wmma_f32_16x16x32_bf16(
            false, afrag[kb], false, bfR, (short)0, accR[tI], false, false);
      }
    }

    // Phase C: RG-LRU gating (f32) -> a, normalized_x into LDS scratch
#pragma unroll
    for (int tI = 0; tI < 2; ++tI) {
      const int cLocal = (wave * 2 + tI) * 16 + (lane & 15);
#pragma unroll
      for (int v = 0; v < 8; ++v) {
        float ig = 1.0f / (1.0f + __expf(-(accI[tI][v] + bI[tI])));
        float rg = 1.0f / (1.0f + __expf(-(accR[tI][v] + bR[tI])));
        float log_a = -8.0f * rg * sp[tI];
        float a  = __expf(log_a);
        float a2 = __expf(2.0f * log_a);
        float m  = sqrtf(fmaxf(1.0f - a2, 1e-10f));
        if ((rstm >> v) & 1) { m = 1.0f; a = 0.0f; }
        const int r = rbase + v;
        aS [r * 128 + cLocal] = a;
        xnS[r * 128 + cLocal] = xv[tI][v] * ig * m;
      }
    }
    __syncthreads();

    // Phase D: diagonal recurrence, thread = channel, carry in register
#pragma unroll
    for (int r = 0; r < 16; ++r) {
      state = fmaf(aS[r * 128 + tid], state, xnS[r * 128 + tid]);
      outBase[(size_t)(s0 + r) * DMODEL + tid] = state;  // coalesced 512B rows
    }
    __syncthreads();
  }
}

extern "C" void kernel_launch(void* const* d_in, const int* in_sizes, int n_in,
                              void* d_out, int out_size, void* d_ws, size_t ws_size,
                              hipStream_t stream) {
  (void)in_sizes; (void)n_in; (void)out_size; (void)ws_size;
  const float* act = (const float*)d_in[0];
  const int*   pos = (const int*)d_in[1];
  const float* lam = (const float*)d_in[2];
  const float* wI  = (const float*)d_in[3];
  const float* bI  = (const float*)d_in[4];
  const float* wR  = (const float*)d_in[5];
  const float* bR  = (const float*)d_in[6];
  float* out = (float*)d_out;
  unsigned int* wPrep = (unsigned int*)d_ws;   // 2.62 MB of fragment-layout bf16 weights

  rglru_prep_weights<<<320, 256, 0, stream>>>(wI, wR, wPrep);
  dim3 grid(2, HNUM, BATCH);
  rglru_fused<<<grid, 128, 0, stream>>>(act, pos, lam, bI, bR, wPrep, out);
}
